// LanguageModel_79164837199943
// MI455X (gfx1250) — matrix operand order repaired and checked by reading, not compiled, not run on hardware
//
#include <hip/hip_runtime.h>
#include <hip/hip_bf16.h>

// ---------------------------------------------------------------------------
// Problem constants (from the reference)
// ---------------------------------------------------------------------------
#define Lp 512
#define Np 64
#define Kin 38
#define Ep 1024
#define Hp 1024
#define H3 3072
#define Cc 38
#define ROWS (Lp * Np)          // 32768
#define NLAYERS 3

typedef __bf16 bf16_t;
typedef __attribute__((ext_vector_type(16))) __bf16 v16bf;
typedef __attribute__((ext_vector_type(8)))  __bf16 v8bf;
typedef __attribute__((ext_vector_type(8)))  float  v8f;

#define CAT16(lo, hi) __builtin_shufflevector((lo), (hi), 0,1,2,3,4,5,6,7,8,9,10,11,12,13,14,15)

__device__ __forceinline__ v8f wmma_bf16(v16bf a, v16bf b, v8f c) {
    // (neg_a, A, neg_b, B, c_mod, C, reuse_a, reuse_b)
    return __builtin_amdgcn_wmma_f32_16x16x32_bf16(false, a, false, b, (short)0, c, false, false);
}

__device__ __forceinline__ bf16_t f2bf(float f) {
    unsigned u = __builtin_bit_cast(unsigned, f);
    u += 0x7FFFu + ((u >> 16) & 1u);            // round-to-nearest-even
    unsigned short h = (unsigned short)(u >> 16);
    return __builtin_bit_cast(bf16_t, h);
}

__device__ __forceinline__ float sigmoidf_(float x) {
    return 1.0f / (1.0f + __expf(-x));
}

// ---------------------------------------------------------------------------
// f32 -> bf16 conversion (weights, activations)
// ---------------------------------------------------------------------------
__global__ void __launch_bounds__(256)
k_f32_to_bf16(const float* __restrict__ src, bf16_t* __restrict__ dst, int n) {
    for (int i = blockIdx.x * 256 + threadIdx.x; i < n; i += gridDim.x * 256)
        dst[i] = f2bf(src[i]);
}

// ---------------------------------------------------------------------------
// Embedding: x[row, e] = inputs[row, 0:38] . emb_W[e, 0:38] + emb_b[e]
// ---------------------------------------------------------------------------
__global__ void __launch_bounds__(128)
k_embed(const float* __restrict__ in, const float* __restrict__ W,
        const float* __restrict__ b, bf16_t* __restrict__ xbf) {
    __shared__ float s[Kin];
    const int row = blockIdx.x;
    if (threadIdx.x < Kin) s[threadIdx.x] = in[(size_t)row * Kin + threadIdx.x];
    __syncthreads();
    for (int e = threadIdx.x; e < Ep; e += 128) {
        const float* w = W + (size_t)e * Kin;
        float acc = b[e];
#pragma unroll
        for (int k = 0; k < Kin; ++k) acc += s[k] * w[k];
        xbf[(size_t)row * Ep + e] = f2bf(acc);
    }
}

// ---------------------------------------------------------------------------
// Input projection GEMM: xg (ROWS x 3H) = X_bf16 (ROWS x E) @ W_bf16^T + b_ih
// grid = (3H/128, ROWS/128); block = 256 threads = 8 waves.
// Block tile 128M x 128N; B k-chunk (16KB) staged through LDS with a one-chunk
// software pipeline. Per k-step: all 8 B tiles are pulled from LDS into
// distinct registers, a sched_barrier pins them there, then 8 WMMAs issue
// back-to-back with staggered dscnt waits.
// ---------------------------------------------------------------------------
#define XP_NT 8                  // n-tiles per block (128 cols)
__global__ void __launch_bounds__(256)
k_xproj(const bf16_t* __restrict__ X, const bf16_t* __restrict__ W,
        const float* __restrict__ bias, float* __restrict__ out) {
    __shared__ __align__(32) bf16_t sB[2 * XP_NT * 512];   // 16 tiles * 1KB = 16KB
    const int tid  = threadIdx.x;
    const int wave = tid >> 5;
    const int lane = tid & 31;
    const int half = lane >> 4;                 // 0 | 1
    const int l15  = lane & 15;
    const int mbase = blockIdx.y * 128 + wave * 16;
    const int nbase = blockIdx.x * 128;
    const int loff  = lane * 8;

    v8f acc[XP_NT];
#pragma unroll
    for (int i = 0; i < XP_NT; ++i) acc[i] = v8f{};

    const bf16_t* Arow = X + (size_t)(mbase + l15) * Ep + (half ? 8 : 0);

    // each thread stages 2 slots (slot = tile*32 + lane-in-tile, 32B each)
    v8bf s_lo[2], s_hi[2];
    auto stage_load = [&](int k0) {
#pragma unroll
        for (int s = 0; s < 2; ++s) {
            const int sidx = tid + s * 256;
            const int tile = sidx >> 5;         // 0..15  (= ntile*2 + kk)
            const int ln   = sidx & 31;
            const int nt   = tile >> 1;
            const int kk   = tile & 1;
            const int row  = nbase + nt * 16 + (ln & 15);
            const int koff = k0 + kk * 32 + ((ln >> 4) << 3);
            const bf16_t* g = W + (size_t)row * Ep + koff;
            s_lo[s] = *(const v8bf*)g;
            s_hi[s] = *(const v8bf*)(g + 16);
        }
    };
    auto stage_store = [&]() {
#pragma unroll
        for (int s = 0; s < 2; ++s) {
            const int sidx = tid + s * 256;
            const int tile = sidx >> 5;
            const int ln   = sidx & 31;
            bf16_t* d = sB + tile * 512;
            *(v8bf*)(d + ln * 8)       = s_lo[s];
            *(v8bf*)(d + 256 + ln * 8) = s_hi[s];
        }
    };

    stage_load(0);
    for (int c = 0; c < Ep / 64; ++c) {
        stage_store();
        __syncthreads();
        if (c < Ep / 64 - 1) stage_load((c + 1) * 64);   // overlap with compute
        const int k0 = c * 64;
#pragma unroll
        for (int kk = 0; kk < 2; ++kk) {
            const bf16_t* ap = Arow + k0 + kk * 32;
            v8bf alo = *(const v8bf*)ap;                 // global A first (long latency)
            v8bf ahi = *(const v8bf*)(ap + 16);
            v16bf bm[XP_NT];
#pragma unroll
            for (int nt = 0; nt < XP_NT; ++nt) {
                const bf16_t* d = sB + (nt * 2 + kk) * 512;
                bm[nt] = CAT16(*(const v8bf*)(d + loff),
                               *(const v8bf*)(d + 256 + loff));
            }
            __builtin_amdgcn_sched_barrier(0);           // pin loads before WMMAs
            v16bf a = CAT16(alo, ahi);
#pragma unroll
            for (int nt = 0; nt < XP_NT; ++nt)
                acc[nt] = wmma_bf16(a, bm[nt], acc[nt]);
        }
        __syncthreads();
    }
#pragma unroll
    for (int nt = 0; nt < XP_NT; ++nt) {
        const int col = nbase + nt * 16 + l15;
        const float bv = bias[col];
#pragma unroll
        for (int i = 0; i < 8; ++i) {
            const int row = mbase + i + 8 * half;
            out[(size_t)row * H3 + col] = acc[nt][i] + bv;
        }
    }
}

// ---------------------------------------------------------------------------
// init: zero h (read buffer) and barrier words
// ---------------------------------------------------------------------------
__global__ void __launch_bounds__(256)
k_init_state(float* hf0, bf16_t* hbf0, int* bar) {
    const int i = blockIdx.x * 256 + threadIdx.x;     // 256 blocks -> 65536
    if (i < Np * Hp) { hf0[i] = 0.0f; hbf0[i] = __builtin_bit_cast(bf16_t, (unsigned short)0); }
    if (i == 0) { bar[0] = 0; bar[1] = 0; }
}

// ---------------------------------------------------------------------------
// device-scope grid barrier (64 co-resident blocks)
// ---------------------------------------------------------------------------
__device__ __forceinline__ void grid_sync(int* bar, int nblk) {
    __syncthreads();
    if (threadIdx.x == 0) {
        __threadfence();
        int g = __hip_atomic_load(bar + 1, __ATOMIC_RELAXED, __HIP_MEMORY_SCOPE_AGENT);
        if (__hip_atomic_fetch_add(bar, 1, __ATOMIC_ACQ_REL, __HIP_MEMORY_SCOPE_AGENT) == nblk - 1) {
            __hip_atomic_store(bar, 0, __ATOMIC_RELAXED, __HIP_MEMORY_SCOPE_AGENT);
            __hip_atomic_fetch_add(bar + 1, 1, __ATOMIC_RELEASE, __HIP_MEMORY_SCOPE_AGENT);
        } else {
            while (__hip_atomic_load(bar + 1, __ATOMIC_ACQUIRE, __HIP_MEMORY_SCOPE_AGENT) == g)
                __builtin_amdgcn_s_sleep(1);
        }
        __threadfence();
    }
    __syncthreads();
}

// ---------------------------------------------------------------------------
// Persistent GRU layer: 64 blocks x 128 threads (4 waves, ~1 wave/SIMD ->
// pure ILP, so the B/A operands are explicitly double-buffered in registers).
// block b owns H-columns [16b, 16b+16); wave w owns batch rows [16w, 16w+16).
// w_hh panel (96KB) staged ONCE into dynamic LDS. Per k-step, iteration ks
// issues the 6 LDS + 2 global loads for ks+1 into the alternate buffer
// (sched_barrier keeps them there), then runs the 3 WMMAs of ks on operands
// loaded a full iteration earlier. xg / h_prev gate operands prefetched at
// the top of each step. h double-buffered, grid barrier per step.
// ---------------------------------------------------------------------------
__global__ void __launch_bounds__(128)
k_gru_layer(const float* __restrict__ xg, const bf16_t* __restrict__ whh,
            const float* __restrict__ bhh,
            float* hf0, float* hf1, bf16_t* hbf0, bf16_t* hbf1,
            float* __restrict__ y, int* bar) {
    extern __shared__ __align__(32) char smem_raw[];
    bf16_t* sW = (bf16_t*)smem_raw;              // 96 tiles * 512 elems = 96KB

    const int wave = threadIdx.x >> 5;
    const int lane = threadIdx.x & 31;
    const int half = lane >> 4;
    const int l15  = lane & 15;
    const int mbase = wave * 16;                 // batch rows
    const int j     = blockIdx.x * 16 + l15;     // my H column (C-layout N)
    const int loff  = lane * 8;

    // ---- stage w_hh panel into LDS (once) ----
    for (int tile = wave; tile < 96; tile += 4) {
        const int gate = tile >> 5;              // 0..2
        const int ks   = tile & 31;              // k-step
        const int row  = gate * Hp + blockIdx.x * 16 + l15;
        const int koff = ks * 32 + (half << 3);
        const bf16_t* g = whh + (size_t)row * Hp + koff;
        v8bf lo = *(const v8bf*)g;
        v8bf hi = *(const v8bf*)(g + 16);
        bf16_t* d = sW + tile * 512;
        *(v8bf*)(d + loff)       = lo;
        *(v8bf*)(d + 256 + loff) = hi;
    }
    __syncthreads();

    const float bR = bhh[j];
    const float bZ = bhh[Hp + j];
    const float bN = bhh[2 * Hp + j];

    for (int t = 0; t < Lp; ++t) {
        const bf16_t* hbr = (t & 1) ? hbf1 : hbf0;   // read h_{t-1}
        bf16_t*       hbw = (t & 1) ? hbf0 : hbf1;   // write h_t
        const float*  hfr = (t & 1) ? hf1 : hf0;
        float*        hfw = (t & 1) ? hf0 : hf1;
        const float*  xgt = xg + (size_t)t * Np * H3;

        // prefetch gate operands (independent of the recurrent GEMM)
        float xr[8], xz[8], xn[8], hp[8];
#pragma unroll
        for (int i = 0; i < 8; ++i) {
            const int mg = mbase + i + 8 * half;
            const float* xrow = xgt + (size_t)mg * H3;
            xr[i] = xrow[j];
            xz[i] = xrow[Hp + j];
            xn[i] = xrow[2 * Hp + j];
            hp[i] = hfr[(size_t)mg * Hp + j];
        }

        v8f aR = {}, aZ = {}, aN = {};
        const bf16_t* Arow = hbr + (size_t)(mbase + l15) * Hp + (half ? 8 : 0);

        // double-buffered operands; preload pipeline stage for ks = 0
        v16bf bRb[2], bZb[2], bNb[2];
        v8bf alo[2], ahi[2];
        {
            const bf16_t* dR = sW + 0 * 512;
            const bf16_t* dZ = sW + 32 * 512;
            const bf16_t* dN = sW + 64 * 512;
            bRb[0] = CAT16(*(const v8bf*)(dR + loff), *(const v8bf*)(dR + 256 + loff));
            bZb[0] = CAT16(*(const v8bf*)(dZ + loff), *(const v8bf*)(dZ + 256 + loff));
            bNb[0] = CAT16(*(const v8bf*)(dN + loff), *(const v8bf*)(dN + 256 + loff));
            alo[0] = *(const v8bf*)(Arow);
            ahi[0] = *(const v8bf*)(Arow + 16);
        }
#pragma unroll 2
        for (int ks = 0; ks < 32; ++ks) {
            const int cur = ks & 1;
            const int nxt = cur ^ 1;
            const int ksn = (ks + 1) & 31;           // wraps harmlessly on last iter
            // issue next iteration's loads into the alternate buffer
            const bf16_t* dR = sW + ksn * 512;
            const bf16_t* dZ = sW + (32 + ksn) * 512;
            const bf16_t* dN = sW + (64 + ksn) * 512;
            bRb[nxt] = CAT16(*(const v8bf*)(dR + loff), *(const v8bf*)(dR + 256 + loff));
            bZb[nxt] = CAT16(*(const v8bf*)(dZ + loff), *(const v8bf*)(dZ + 256 + loff));
            bNb[nxt] = CAT16(*(const v8bf*)(dN + loff), *(const v8bf*)(dN + 256 + loff));
            alo[nxt] = *(const v8bf*)(Arow + ksn * 32);
            ahi[nxt] = *(const v8bf*)(Arow + ksn * 32 + 16);
            __builtin_amdgcn_sched_barrier(0);       // keep prefetch ahead of WMMAs
            v16bf a = CAT16(alo[cur], ahi[cur]);
            aR = wmma_bf16(a, bRb[cur], aR);
            aZ = wmma_bf16(a, bZb[cur], aZ);
            aN = wmma_bf16(a, bNb[cur], aN);
        }

#pragma unroll
        for (int i = 0; i < 8; ++i) {
            const int mg = mbase + i + 8 * half;             // batch row
            const float r  = sigmoidf_(xr[i] + aR[i] + bR);
            const float z  = sigmoidf_(xz[i] + aZ[i] + bZ);
            const float nn = tanhf    (xn[i] + r * (aN[i] + bN));
            const float hnew = (1.0f - z) * nn + z * hp[i];
            y[((size_t)t * Np + mg) * Hp + j] = hnew;
            hfw[(size_t)mg * Hp + j] = hnew;
            hbw[(size_t)mg * Hp + j] = f2bf(hnew);
        }
        grid_sync(bar, gridDim.x);
    }
}

// ---------------------------------------------------------------------------
// Logits: out[row, c] = y[row, :] . out_W[c, :] + out_b[c]
// ---------------------------------------------------------------------------
__global__ void __launch_bounds__(128)
k_logits(const float* __restrict__ y, const float* __restrict__ W,
         const float* __restrict__ b, float* __restrict__ out) {
    __shared__ float s[Hp];
    const size_t row = blockIdx.x;
    for (int i = threadIdx.x; i < Hp; i += 128) s[i] = y[row * Hp + i];
    __syncthreads();
    if (threadIdx.x < Cc) {
        const float* w = W + (size_t)threadIdx.x * Hp;
        float acc = b[threadIdx.x];
#pragma unroll 4
        for (int k = 0; k < Hp; ++k) acc += s[k] * w[k];
        out[row * Cc + threadIdx.x] = acc;
    }
}

// ---------------------------------------------------------------------------
// host launcher
// ---------------------------------------------------------------------------
extern "C" void kernel_launch(void* const* d_in, const int* in_sizes, int n_in,
                              void* d_out, int out_size, void* d_ws, size_t ws_size,
                              hipStream_t stream) {
    const float* inputs = (const float*)d_in[0];   // (L, N, 38)
    const float* emb_W  = (const float*)d_in[1];   // (E, 38)
    const float* emb_b  = (const float*)d_in[2];   // (E,)
    const float* w_ih   = (const float*)d_in[3];   // (3, 3H, E)
    const float* w_hh   = (const float*)d_in[4];   // (3, 3H, H)
    const float* b_ih   = (const float*)d_in[5];   // (3, 3H)
    const float* b_hh   = (const float*)d_in[6];   // (3, 3H)
    const float* out_W  = (const float*)d_in[7];   // (C, H)
    const float* out_b  = (const float*)d_in[8];   // (C,)
    float* out = (float*)d_out;

    char* ws = (char*)d_ws;
    size_t off = 0;
    auto alloc = [&](size_t bytes) -> void* {
        void* p = ws + off;
        off += (bytes + 255) & ~(size_t)255;
        return p;
    };

    const size_t WIH = (size_t)NLAYERS * H3 * Ep;
    const size_t WHH = (size_t)NLAYERS * H3 * Hp;

    bf16_t* wih_bf = (bf16_t*)alloc(WIH * sizeof(bf16_t));
    bf16_t* whh_bf = (bf16_t*)alloc(WHH * sizeof(bf16_t));
    bf16_t* xbf    = (bf16_t*)alloc((size_t)ROWS * Ep * sizeof(bf16_t));
    float*  xg     = (float*)alloc((size_t)ROWS * H3 * sizeof(float));
    float*  yb     = (float*)alloc((size_t)ROWS * Hp * sizeof(float));
    float*  hf0    = (float*)alloc((size_t)Np * Hp * sizeof(float));
    float*  hf1    = (float*)alloc((size_t)Np * Hp * sizeof(float));
    bf16_t* hbf0   = (bf16_t*)alloc((size_t)Np * Hp * sizeof(bf16_t));
    bf16_t* hbf1   = (bf16_t*)alloc((size_t)Np * Hp * sizeof(bf16_t));
    int*    bar    = (int*)alloc(256);

    // weights -> bf16
    k_f32_to_bf16<<<4096, 256, 0, stream>>>(w_ih, wih_bf, (int)WIH);
    k_f32_to_bf16<<<4096, 256, 0, stream>>>(w_hh, whh_bf, (int)WHH);

    // embedding -> bf16 activations
    k_embed<<<ROWS, 128, 0, stream>>>(inputs, emb_W, emb_b, xbf);

    for (int l = 0; l < NLAYERS; ++l) {
        const bf16_t* wih_l = wih_bf + (size_t)l * H3 * Ep;
        const bf16_t* whh_l = whh_bf + (size_t)l * H3 * Hp;
        const float*  bih_l = b_ih + (size_t)l * H3;
        const float*  bhh_l = b_hh + (size_t)l * H3;

        // xg = x @ w_ih^T + b_ih   (WMMA, LDS-staged + pipelined B)
        k_xproj<<<dim3(H3 / 128, ROWS / 128), 256, 0, stream>>>(xbf, wih_l, bih_l, xg);

        // h0 = 0, barrier reset
        k_init_state<<<256, 256, 0, stream>>>(hf0, hbf0, bar);

        // persistent recurrent scan (WMMA per step, w_hh panel in 96KB LDS)
        k_gru_layer<<<Hp / 16, 128, 96 * 1024, stream>>>(xg, whh_l, bhh_l,
                                                         hf0, hf1, hbf0, hbf1, yb, bar);

        // layer output becomes next layer's bf16 input
        if (l < NLAYERS - 1)
            k_f32_to_bf16<<<8192, 256, 0, stream>>>(yb, xbf, ROWS * Ep);
    }

    // logits
    k_logits<<<ROWS, 128, 0, stream>>>(yb, out_W, out_b, out);
}
